// GraphAttentionNetwork_14912126452048
// MI455X (gfx1250) — matrix operand order repaired and checked by reading, not compiled, not run on hardware
//
#include <hip/hip_runtime.h>
#include <hip/hip_bf16.h>
#include <cstdint>

typedef __attribute__((ext_vector_type(16))) _Float16 v16h;
typedef __attribute__((ext_vector_type(8)))  _Float16 v8h;
typedef __attribute__((ext_vector_type(8)))  float    v8f;

#define NHEADS 8
#define CDIM   32
#define FIN    128
#define FMID   256   // NHEADS * CDIM
#define NEG_SLOPE 0.2f
#define EPS 1e-16f

// ---- order-preserving float <-> uint map (for atomicMax on floats) ----
__device__ __forceinline__ unsigned ford(float f) {
  unsigned u = __float_as_uint(f);
  return (u & 0x80000000u) ? ~u : (u | 0x80000000u);
}
__device__ __forceinline__ float unford(unsigned u) {
  unsigned v = (u & 0x80000000u) ? (u & 0x7FFFFFFFu) : ~u;
  return __uint_as_float(v);
}
__device__ __forceinline__ float leaky(float v) { return v > 0.f ? v : NEG_SLOPE * v; }

// ---------------- precision conversion ----------------
__global__ void cvt_x_f16(const float* __restrict__ x, _Float16* __restrict__ xf, int total) {
  int i = blockIdx.x * blockDim.x + threadIdx.x;
  if (i < total) xf[i] = (_Float16)x[i];
}
// W1 [128,256] row-major -> W1t [256,128] f16 (so B fragments are contiguous)
__global__ void cvt_w1_f16t(const float* __restrict__ W1, _Float16* __restrict__ Wt) {
  int i = blockIdx.x * blockDim.x + threadIdx.x;
  if (i >= FIN * FMID) return;
  int k = i / FMID, n = i % FMID;
  Wt[n * FIN + k] = (_Float16)W1[i];
}

// ---------------- WMMA GEMM: h1[M,256] = x_f16[M,128] @ W1[128,256] ----------------
__global__ void gemm_wmma(const _Float16* __restrict__ A, const _Float16* __restrict__ Bt,
                          float* __restrict__ C, int M) {
  int wave = (int)((blockIdx.x * blockDim.x + threadIdx.x) >> 5);
  int lane = threadIdx.x & 31;
  int tileN = wave % (FMID / 16);
  int tileM = wave / (FMID / 16);
  if (tileM * 16 >= M) return;              // wave-uniform guard (EXEC stays full)

  int ln16 = lane & 15;                     // M index for A frag, N index for B frag
  int kg   = lane >> 4;                     // K-group select
  const _Float16* arow = A  + (size_t)(tileM * 16 + ln16) * FIN;
  const _Float16* brow = Bt + (size_t)(tileN * 16 + ln16) * FIN;

  v8f c = {};
#pragma unroll
  for (int kb = 0; kb < 4; ++kb) {
    int base = kb * 32;
    // A 16x32 f16: half i -> K = kg*16 + i
    v8h alo = *(const v8h*)(arow + base + kg * 16);
    v8h ahi = *(const v8h*)(arow + base + kg * 16 + 8);
    v16h a = __builtin_shufflevector(alo, ahi, 0,1,2,3,4,5,6,7,8,9,10,11,12,13,14,15);
    // B 32x16 f16: half i -> K = kg*16 + i, column = lane&15 (contiguous in W1t row)
    v16h b = *(const v16h*)(brow + base + kg * 16);
    c = __builtin_amdgcn_wmma_f32_16x16x32_f16(false, a, false, b, (short)0, c, false, false);
  }
  // D layout: VGPR r -> row = r + (lane<16 ? 0 : 8), col = lane&15
  int row0 = tileM * 16 + kg * 8;
  int col  = tileN * 16 + ln16;
#pragma unroll
  for (int r = 0; r < 8; ++r)
    C[(size_t)(row0 + r) * FMID + col] = c[r];
}

// ---------------- attention coefficients: as1/ad1 [N,8] ----------------
__global__ void attn_coef(const float* __restrict__ h1, const float* __restrict__ a_src,
                          const float* __restrict__ a_dst, float* __restrict__ as1,
                          float* __restrict__ ad1, int N) {
  int t = blockIdx.x * blockDim.x + threadIdx.x;
  if (t >= N * NHEADS) return;
  int n = t >> 3, h = t & 7;
  const float* hp  = h1 + (size_t)n * FMID + h * CDIM;
  const float* asp = a_src + h * CDIM;
  const float* adp = a_dst + h * CDIM;
  float s0 = 0.f, s1 = 0.f;
#pragma unroll
  for (int c = 0; c < CDIM; ++c) { float v = hp[c]; s0 += v * asp[c]; s1 += v * adp[c]; }
  as1[t] = s0; ad1[t] = s1;
}

// ---------------- layer-1 edge passes ----------------
__global__ void edge_max1(const int* __restrict__ src, const int* __restrict__ dst, int E, int N,
                          const float* __restrict__ as1, const float* __restrict__ ad1,
                          unsigned* __restrict__ m1) {
  int e = blockIdx.x * blockDim.x + threadIdx.x;
  if (e >= E + N) return;
  int s = (e < E) ? src[e] : (e - E);
  int d = (e < E) ? dst[e] : (e - E);
#pragma unroll
  for (int h = 0; h < NHEADS; ++h) {
    float v = leaky(as1[s * NHEADS + h] + ad1[d * NHEADS + h]);
    atomicMax(&m1[d * NHEADS + h], ford(v));
  }
}

__global__ void edge_sum1(const int* __restrict__ src, const int* __restrict__ dst, int E, int N,
                          const float* __restrict__ as1, const float* __restrict__ ad1,
                          const unsigned* __restrict__ m1, float* __restrict__ sum1) {
  int e = blockIdx.x * blockDim.x + threadIdx.x;
  if (e >= E + N) return;
  int s = (e < E) ? src[e] : (e - E);
  int d = (e < E) ? dst[e] : (e - E);
#pragma unroll
  for (int h = 0; h < NHEADS; ++h) {
    int dh = d * NHEADS + h;
    float v = leaky(as1[s * NHEADS + h] + ad1[dh]);
    atomicAdd(&sum1[dh], __expf(v - unford(m1[dh])));
  }
}

// one block per edge: 256 threads = 8 heads x 32 channels (coalesced 1KB gather)
__global__ void edge_aggr1(const int* __restrict__ src, const int* __restrict__ dst, int E,
                           const float* __restrict__ as1, const float* __restrict__ ad1,
                           const unsigned* __restrict__ m1, const float* __restrict__ sum1,
                           const float* __restrict__ h1, float* __restrict__ out1) {
  int e = blockIdx.x;
  int s = (e < E) ? src[e] : (e - E);
  int d = (e < E) ? dst[e] : (e - E);
  int h = threadIdx.x >> 5;
  int c = threadIdx.x & 31;
  int dh = d * NHEADS + h;
  float ev    = leaky(as1[s * NHEADS + h] + ad1[dh]);
  float ex    = __expf(ev - unford(m1[dh]));
  float alpha = ex / (sum1[dh] + EPS);
  float v = h1[(size_t)s * FMID + h * CDIM + c] * alpha;
  atomicAdd(&out1[(size_t)d * FMID + h * CDIM + c], v);
}

// ---------------- layer-2 node transform: h2[n] = sum_k elu(out1+bias1)*W2 ----------------
__global__ void node2(const float* __restrict__ out1, const float* __restrict__ bias1,
                      const float* __restrict__ W2, float* __restrict__ h2, int N) {
  int wave = (int)((blockIdx.x * blockDim.x + threadIdx.x) >> 5);
  int lane = threadIdx.x & 31;
  if (wave >= N) return;
  float acc = 0.f;
#pragma unroll
  for (int j = 0; j < FMID / 32; ++j) {
    int k = j * 32 + lane;
    float v = out1[(size_t)wave * FMID + k] + bias1[k];
    v = v > 0.f ? v : (__expf(v) - 1.f);   // ELU
    acc += v * W2[k];
  }
#pragma unroll
  for (int off = 16; off > 0; off >>= 1) acc += __shfl_xor(acc, off, 32);
  if (lane == 0) h2[wave] = acc;
}

// ---------------- layer-2 edge passes (scalar per edge) ----------------
__global__ void edge_max2(const int* __restrict__ src, const int* __restrict__ dst, int E, int N,
                          const float* __restrict__ h2, const float* __restrict__ a_s2,
                          const float* __restrict__ a_d2, unsigned* __restrict__ m2) {
  int e = blockIdx.x * blockDim.x + threadIdx.x;
  if (e >= E + N) return;
  int s = (e < E) ? src[e] : (e - E);
  int d = (e < E) ? dst[e] : (e - E);
  float v = leaky(h2[s] * a_s2[0] + h2[d] * a_d2[0]);
  atomicMax(&m2[d], ford(v));
}

__global__ void edge_sum2(const int* __restrict__ src, const int* __restrict__ dst, int E, int N,
                          const float* __restrict__ h2, const float* __restrict__ a_s2,
                          const float* __restrict__ a_d2, const unsigned* __restrict__ m2,
                          float* __restrict__ sum2) {
  int e = blockIdx.x * blockDim.x + threadIdx.x;
  if (e >= E + N) return;
  int s = (e < E) ? src[e] : (e - E);
  int d = (e < E) ? dst[e] : (e - E);
  float v = leaky(h2[s] * a_s2[0] + h2[d] * a_d2[0]);
  atomicAdd(&sum2[d], __expf(v - unford(m2[d])));
}

__global__ void edge_aggr2(const int* __restrict__ src, const int* __restrict__ dst, int E, int N,
                           const float* __restrict__ h2, const float* __restrict__ a_s2,
                           const float* __restrict__ a_d2, const unsigned* __restrict__ m2,
                           const float* __restrict__ sum2, float* __restrict__ agg2) {
  int e = blockIdx.x * blockDim.x + threadIdx.x;
  if (e >= E + N) return;
  int s = (e < E) ? src[e] : (e - E);
  int d = (e < E) ? dst[e] : (e - E);
  float v  = leaky(h2[s] * a_s2[0] + h2[d] * a_d2[0]);
  float ex = __expf(v - unford(m2[d]));
  atomicAdd(&agg2[d], h2[s] * (ex / (sum2[d] + EPS)));
}

__global__ void finalize_k(const float* __restrict__ agg2, const float* __restrict__ bias2,
                           float* __restrict__ out, int N) {
  int n = blockIdx.x * blockDim.x + threadIdx.x;
  if (n >= N) return;
  float v = agg2[n] + bias2[0];
  out[n] = 1.f / (1.f + __expf(-v));
}

// ---------------- host launcher ----------------
extern "C" void kernel_launch(void* const* d_in, const int* in_sizes, int n_in,
                              void* d_out, int out_size, void* d_ws, size_t ws_size,
                              hipStream_t stream) {
  (void)n_in; (void)out_size; (void)ws_size;
  const float* x     = (const float*)d_in[0];
  const int*   ei    = (const int*)  d_in[1];
  const float* W1    = (const float*)d_in[2];
  const float* asrc1 = (const float*)d_in[3];
  const float* adst1 = (const float*)d_in[4];
  const float* bias1 = (const float*)d_in[5];
  const float* W2    = (const float*)d_in[6];
  const float* asrc2 = (const float*)d_in[7];
  const float* adst2 = (const float*)d_in[8];
  const float* bias2 = (const float*)d_in[9];

  const int N = in_sizes[0] / FIN;
  const int E = in_sizes[1] / 2;
  const int* src = ei;
  const int* dst = ei + E;
  const int tot = E + N;

  // workspace layout: zero-init buffers first (one contiguous memset)
  char* ws = (char*)d_ws;
  size_t off = 0;
  auto carve = [&](size_t bytes) -> char* {
    char* p = ws + off;
    off += (bytes + 255) & ~(size_t)255;
    return p;
  };
  unsigned* m1   = (unsigned*)carve((size_t)N * NHEADS * 4);
  float*    sum1 = (float*)   carve((size_t)N * NHEADS * 4);
  float*    out1 = (float*)   carve((size_t)N * FMID * 4);
  unsigned* m2   = (unsigned*)carve((size_t)N * 4);
  float*    sum2 = (float*)   carve((size_t)N * 4);
  float*    agg2 = (float*)   carve((size_t)N * 4);
  size_t zbytes = off;
  _Float16* xf  = (_Float16*)carve((size_t)N * FIN * 2);
  _Float16* w1t = (_Float16*)carve((size_t)FMID * FIN * 2);
  float*    h1  = (float*)   carve((size_t)N * FMID * 4);
  float*    as1 = (float*)   carve((size_t)N * NHEADS * 4);
  float*    ad1 = (float*)   carve((size_t)N * NHEADS * 4);
  float*    h2  = (float*)   carve((size_t)N * 4);

  hipMemsetAsync(d_ws, 0, zbytes, stream);

  cvt_x_f16  <<<(N * FIN + 255) / 256, 256, 0, stream>>>(x, xf, N * FIN);
  cvt_w1_f16t<<<(FIN * FMID + 255) / 256, 256, 0, stream>>>(W1, w1t);

  int waves = ((N + 15) / 16) * (FMID / 16);
  gemm_wmma<<<(waves + 7) / 8, 256, 0, stream>>>(xf, w1t, h1, N);

  attn_coef<<<(N * NHEADS + 255) / 256, 256, 0, stream>>>(h1, asrc1, adst1, as1, ad1, N);

  edge_max1 <<<(tot + 255) / 256, 256, 0, stream>>>(src, dst, E, N, as1, ad1, m1);
  edge_sum1 <<<(tot + 255) / 256, 256, 0, stream>>>(src, dst, E, N, as1, ad1, m1, sum1);
  edge_aggr1<<<tot, 256, 0, stream>>>(src, dst, E, as1, ad1, m1, sum1, h1, out1);

  node2<<<((N * 32) + 255) / 256, 256, 0, stream>>>(out1, bias1, W2, h2, N);

  edge_max2 <<<(tot + 255) / 256, 256, 0, stream>>>(src, dst, E, N, h2, asrc2, adst2, m2);
  edge_sum2 <<<(tot + 255) / 256, 256, 0, stream>>>(src, dst, E, N, h2, asrc2, adst2, m2, sum2);
  edge_aggr2<<<(tot + 255) / 256, 256, 0, stream>>>(src, dst, E, N, h2, asrc2, adst2, m2, sum2, agg2);

  finalize_k<<<(N + 255) / 256, 256, 0, stream>>>(agg2, bias2, (float*)d_out, N);
}
